// Multihead_Attention_T_Branch_RICAT_54863912239663
// MI455X (gfx1250) — matrix operand order repaired and checked by reading, not compiled, not run on hardware
//
#include <hip/hip_runtime.h>
#include <hip/hip_bf16.h>
#include <stdint.h>
#include <stddef.h>

typedef __attribute__((ext_vector_type(16))) _Float16 v16h;
typedef __attribute__((ext_vector_type(8)))  float    v8f;

#define WMMA_F16(a, b, c) \
  __builtin_amdgcn_wmma_f32_16x16x32_f16(false, (a), false, (b), (short)0, (c), false, false)

constexpr int Ndim = 4, Cdim = 64, Fdim = 256, Tdim = 256, Ddim = 128, Hdim = 32;
constexpr int Bdim = Ndim * Fdim;                 // 1024 sequences
constexpr size_t Mrows = (size_t)Bdim * Tdim;     // 262144 (B*T)

__device__ inline float wave_sum(float v) {
#pragma unroll
  for (int off = 16; off >= 1; off >>= 1) v += __shfl_xor(v, off, 32);
  return v;
}
__device__ inline float sigmoidf(float x) { return 1.0f / (1.0f + __expf(-x)); }

// ---------------------------------------------------------------------------
// Tensor Data Mover: async 1-D f16 tile copy global->LDS (D# per ISA ch.8).
// Guarded: falls back to cooperative vector copy when the builtin is absent.
// ---------------------------------------------------------------------------
#if defined(__AMDGCN__) && __has_builtin(__builtin_amdgcn_tensor_load_to_lds)
#define HAVE_TDM 1
#else
#define HAVE_TDM 0
#endif

#if HAVE_TDM
typedef __attribute__((ext_vector_type(4))) unsigned int u32x4;
typedef __attribute__((ext_vector_type(8))) int          i32x8;
typedef __attribute__((ext_vector_type(4))) int          i32x4;

// Copy nelem f16 (nelem <= 8192 so tile_dim0 fits 16 bits) from gptr to the
// LDS byte offset lds_off. Issued once per calling wave; completion tracked
// with TENSORcnt.
__device__ inline void tdm_load_f16_1d(uint32_t lds_off, const void* gptr,
                                       uint32_t nelem) {
  uint64_t ga = (uint64_t)(uintptr_t)gptr;
  u32x4 g0 = {0u, 0u, 0u, 0u};
  g0[0] = 1u;                                   // count=1, user descriptor
  g0[1] = lds_off;                              // lds_addr [63:32]
  g0[2] = (uint32_t)(ga & 0xFFFFFFFFu);         // global_addr[95:64] = ga[31:0]
  g0[3] = (uint32_t)((ga >> 32) & 0x01FFFFFFu)  // global_addr[120:96]
          | (2u << 30);                         // type=2 ("image") [127:126]
  i32x8 g1 = {0, 0, 0, 0, 0, 0, 0, 0};
  g1[0] = (int)(1u << 16);                      // data_size=1 (2 bytes)
  g1[1] = (int)((nelem & 0xFFFFu) << 16);       // tensor_dim0 lo16 @bits[63:48]
  g1[2] = (int)((nelem >> 16) & 0xFFFFu)        // tensor_dim0 hi16 @bits[79:64]
          | (int)(1u << 16);                    // tensor_dim1 = 1 @bits[111:80]
  g1[3] = (int)((nelem & 0xFFFFu) << 16);       // tile_dim0 @bits[127:112]
  g1[4] = 1;                                    // tile_dim1=1, tile_dim2=0
  g1[5] = (int)nelem;                           // tensor_dim0_stride lo32
  g1[6] = 0;
  g1[7] = 0;
  i32x4 z4 = {0, 0, 0, 0};
#if __clang_major__ >= 23
  i32x8 z8 = {0, 0, 0, 0, 0, 0, 0, 0};
  __builtin_amdgcn_tensor_load_to_lds(g0, g1, z4, z4, z8, 0);
#else
  __builtin_amdgcn_tensor_load_to_lds(g0, g1, z4, z4, 0);
#endif
}
#endif  // HAVE_TDM

// ---------------------------------------------------------------------------
// Kernel 0: weight prep — f32 -> f16 weights, fused bias bih+bhh
// ---------------------------------------------------------------------------
__global__ void k_prep(const float* Wq, const float* Wk, const float* Wv,
                       const float* Uq, const float* Uk, const float* Uv,
                       const float* biq, const float* bhq,
                       const float* bik, const float* bhk,
                       const float* biv, const float* bhv,
                       _Float16* __restrict__ wih16,
                       _Float16* __restrict__ whh16,
                       float* __restrict__ bias) {
  int t = blockIdx.x * blockDim.x + threadIdx.x;
  const float* Ws[3] = {Wq, Wk, Wv};
  const float* Us[3] = {Uq, Uk, Uv};
  const float* bis[3] = {biq, bik, biv};
  const float* bhs[3] = {bhq, bhk, bhv};
  if (t < 3 * Ddim * Ddim) {
    int z = t / (Ddim * Ddim), i = t % (Ddim * Ddim);
    wih16[t] = (_Float16)Ws[z][i];
  }
  if (t < 3 * Ddim * Hdim) {
    int z = t / (Ddim * Hdim), i = t % (Ddim * Hdim);
    whh16[t] = (_Float16)Us[z][i];
  }
  if (t < 3 * Ddim) {
    int z = t / Ddim, i = t % Ddim;
    bias[t] = bis[z][i] + bhs[z][i];
  }
}

// ---------------------------------------------------------------------------
// Kernel 1: [N,C,F,T,2] -> LN1 -> xn f16 [B*T, 128]   (one wave per row)
// ---------------------------------------------------------------------------
__global__ void k_ln1(const float* __restrict__ inp, const float* __restrict__ g,
                      const float* __restrict__ bta, _Float16* __restrict__ xn) {
  int lane = threadIdx.x & 31;
  int wv = threadIdx.x >> 5;
  size_t r = (size_t)blockIdx.x * 8 + wv;
  int b = (int)(r / Tdim), t = (int)(r % Tdim);
  int n = b / Fdim, f = b % Fdim;
  float vals[4];
#pragma unroll
  for (int j = 0; j < 4; j++) {
    int c = lane + 32 * j;
    int ch = c & 63, ri = c >> 6;  // first 64 = real, next 64 = imag
    size_t idx = ((((size_t)n * Cdim + ch) * Fdim + f) * Tdim + t) * 2 + ri;
    vals[j] = inp[idx];
  }
  float mean = wave_sum(vals[0] + vals[1] + vals[2] + vals[3]) * (1.0f / Ddim);
  float q = 0.f;
#pragma unroll
  for (int j = 0; j < 4; j++) { float d = vals[j] - mean; q += d * d; }
  float rs = rsqrtf(wave_sum(q) * (1.0f / Ddim) + 1e-5f);
#pragma unroll
  for (int j = 0; j < 4; j++) {
    int c = lane + 32 * j;
    xn[r * Ddim + c] = (_Float16)((vals[j] - mean) * rs * g[c] + bta[c]);
  }
}

// ---------------------------------------------------------------------------
// Kernel 2: Gpre = xn @ Wih^T + (bih+bhh).  M=262144, N=128, K=128.
// Block = 8 waves, M-tile = 64 rows (A via TDM into LDS), each wave owns one
// N-tile of 16, preloads its 4 B-fragments ONCE and reuses them over 4 M
// subtiles -> 16 v_wmma per wave per block.
// ---------------------------------------------------------------------------
__global__ void k_inproj(const _Float16* __restrict__ xn,
                         const _Float16* __restrict__ wih16,
                         const float* __restrict__ bias,
                         float* __restrict__ gpre) {
  __shared__ __attribute__((aligned(16))) _Float16 xa[64 * Ddim];  // 16KB
  int tid = threadIdx.x;
  int lane = tid & 31, wv = tid >> 5;
  int z = blockIdx.y;
  size_t m0 = (size_t)blockIdx.x * 64;
#if HAVE_TDM
  if (wv == 0) {
    tdm_load_f16_1d((uint32_t)(uintptr_t)(void*)xa, xn + m0 * Ddim, 64 * Ddim);
    __builtin_amdgcn_s_wait_tensorcnt(0);
  }
#else
  {
    const uint4* src = (const uint4*)(xn + m0 * Ddim);
    uint4* dst = (uint4*)xa;
#pragma unroll
    for (int i = 0; i < 4; i++) dst[tid + i * 256] = src[tid + i * 256];
  }
#endif
  __builtin_prefetch(xn + (m0 + 64) * Ddim, 0, 1);  // global_prefetch_b8
  __syncthreads();
  int l16 = lane & 15;
  int koff = (lane < 16) ? 0 : 16;
  int kgrp = (lane < 16) ? 0 : 16;
  int voff = (lane < 16) ? 0 : 8;
  int n0 = wv * 16;
  const _Float16* W = wih16 + (size_t)z * Ddim * Ddim;
  float bn = bias[z * Ddim + n0 + l16];
  v16h bf[4];  // B fragments: loaded once, reused for all 4 M subtiles
#pragma unroll
  for (int kc = 0; kc < 4; kc++)
    bf[kc] = *(const v16h*)(W + (size_t)(n0 + l16) * Ddim + kc * 32 + kgrp);
#pragma unroll
  for (int ms = 0; ms < 4; ms++) {
    v8f acc;
#pragma unroll
    for (int e = 0; e < 8; e++) acc[e] = bn;
#pragma unroll
    for (int kc = 0; kc < 4; kc++) {
      int kb = kc * 32;
      v16h a;
#pragma unroll
      for (int i = 0; i < 8; i++) {
        a[i]     = xa[(ms * 16 + l16) * Ddim + kb + koff + i];
        a[8 + i] = xa[(ms * 16 + l16) * Ddim + kb + 8 + koff + i];
      }
      acc = WMMA_F16(a, bf[kc], acc);
    }
#pragma unroll
    for (int v = 0; v < 8; v++) {
      size_t m = m0 + ms * 16 + v + voff;
      gpre[((size_t)z * Mrows + m) * Ddim + n0 + l16] = acc[v];
    }
  }
}

// ---------------------------------------------------------------------------
// Kernel 3: LSTM recurrence. 16 sequences per block, 8 waves cover the 128
// gate outputs. Per step: gates = Gpre + H @ Whh^T via ONE WMMA (K=32).
// ---------------------------------------------------------------------------
__global__ void k_lstm(const float* __restrict__ gpre,
                       const _Float16* __restrict__ whh16,
                       _Float16* __restrict__ qkv) {
  __shared__ __attribute__((aligned(16))) _Float16 hsh[16 * Hdim];
  __shared__ float csh[16 * Hdim];
  __shared__ float gsh[16 * Ddim];
  int tid = threadIdx.x;
  int lane = tid & 31, wv = tid >> 5;
  int z = blockIdx.y;
  int b0 = blockIdx.x * 16;
  hsh[tid] = (_Float16)0.f;  hsh[tid + 256] = (_Float16)0.f;
  csh[tid] = 0.f;            csh[tid + 256] = 0.f;
  __syncthreads();
  int l16 = lane & 15;
  int koff = (lane < 16) ? 0 : 16;
  int kgrp = (lane < 16) ? 0 : 16;
  int voff = (lane < 16) ? 0 : 8;
  int n0 = wv * 16;
  const _Float16* U = whh16 + (size_t)z * Ddim * Hdim;
  v16h bf = *(const v16h*)(U + (size_t)(n0 + l16) * Hdim + kgrp);  // invariant
  for (int t = 0; t < Tdim; t++) {
    v16h a;  // A = current hidden states [16 x 32] from LDS
#pragma unroll
    for (int i = 0; i < 8; i++) {
      a[i]     = hsh[l16 * Hdim + koff + i];
      a[8 + i] = hsh[l16 * Hdim + 8 + koff + i];
    }
    v8f acc;  // C = precomputed input-projection gates (incl. bias)
#pragma unroll
    for (int v = 0; v < 8; v++) {
      size_t m = (size_t)(b0 + v + voff) * Tdim + t;
      acc[v] = gpre[((size_t)z * Mrows + m) * Ddim + n0 + l16];
    }
    acc = WMMA_F16(a, bf, acc);
#pragma unroll
    for (int v = 0; v < 8; v++) gsh[(v + voff) * Ddim + n0 + l16] = acc[v];
    __syncthreads();
#pragma unroll
    for (int rep = 0; rep < 2; rep++) {  // 512 cells, 256 threads
      int idx = tid + rep * 256;
      int bb = idx >> 5, j = idx & 31;
      float gi = gsh[bb * Ddim + j];
      float gf = gsh[bb * Ddim + 32 + j];
      float gg = gsh[bb * Ddim + 64 + j];
      float go = gsh[bb * Ddim + 96 + j];
      float c = sigmoidf(gf) * csh[bb * Hdim + j] + sigmoidf(gi) * tanhf(gg);
      float h = sigmoidf(go) * tanhf(c);
      csh[bb * Hdim + j] = c;
      hsh[bb * Hdim + j] = (_Float16)h;
      qkv[(((size_t)z * Bdim + b0 + bb) * Tdim + t) * Hdim + j] = (_Float16)h;
    }
    __syncthreads();
  }
}

// ---------------------------------------------------------------------------
// Kernel 4: attention. Block = 4 waves, each wave owns 16 query rows of one
// sequence. V is streamed into LDS by the TDM while QK^T (16 WMMA, K=32) and
// softmax run; s_wait_tensorcnt gates the AV phase (8 K-chunks x 2 N-tiles).
// Ref's multiplicative mask is preserved: masked scores are 0, NOT -inf.
// ---------------------------------------------------------------------------
__global__ void k_attn(const _Float16* __restrict__ qkv, float* __restrict__ wvout) {
  __shared__ __attribute__((aligned(16))) _Float16 vsh[Tdim * Hdim];   // 16KB
  __shared__ __attribute__((aligned(16))) _Float16 psh[4][16 * Tdim];  // 32KB
  int tid = threadIdx.x;  // blockDim = 128
  int lane = tid & 31, wv = tid >> 5;
  int b = blockIdx.x >> 2;
  int rows0 = (blockIdx.x & 3) * 64 + wv * 16;
  const _Float16* qb = qkv + ((size_t)0 * Bdim + b) * Tdim * Hdim;
  const _Float16* kb = qkv + ((size_t)1 * Bdim + b) * Tdim * Hdim;
  const _Float16* vb = qkv + ((size_t)2 * Bdim + b) * Tdim * Hdim;
#if HAVE_TDM
  if (wv == 0)  // async: overlaps with QK^T + softmax below
    tdm_load_f16_1d((uint32_t)(uintptr_t)(void*)vsh, vb, Tdim * Hdim);
#else
  {
    const uint4* src = (const uint4*)vb;
    uint4* dst = (uint4*)vsh;
#pragma unroll
    for (int it = 0; it < 8; ++it) dst[tid + it * 128] = src[tid + it * 128];
  }
#endif
  int l16 = lane & 15;
  int koff = (lane < 16) ? 0 : 16;
  int kgrp = (lane < 16) ? 0 : 16;
  int voff = (lane < 16) ? 0 : 8;
  v16h aq;  // Q fragment for this wave's 16 rows (reused across all 16 tiles)
  {
    int tr = rows0 + l16;
#pragma unroll
    for (int i = 0; i < 8; i++) {
      aq[i]     = qb[tr * Hdim + koff + i];
      aq[8 + i] = qb[tr * Hdim + 8 + koff + i];
    }
  }
  for (int nt = 0; nt < 16; ++nt) {  // score tiles along key dim
    v16h bk = *(const v16h*)(kb + (size_t)(nt * 16 + l16) * Hdim + kgrp);
    v8f acc = {};
    acc = WMMA_F16(aq, bk, acc);
#pragma unroll
    for (int v = 0; v < 8; v++) {
      int rl = v + voff;
      int tr = rows0 + rl, tc = nt * 16 + l16;
      float val = (tc <= tr) ? acc[v] * 0.25f : 0.0f;  // energy * tril mask
      psh[wv][rl * Tdim + tc] = (_Float16)val;
    }
  }
  __syncthreads();
  {  // softmax over 256 cols; lane pair (l, l+16) splits each row
    int row = l16;
    int c0 = (lane < 16) ? 0 : 128;
    float mx = -1e30f;
    for (int cc = 0; cc < 128; cc++)
      mx = fmaxf(mx, (float)psh[wv][row * Tdim + c0 + cc]);
    mx = fmaxf(mx, __shfl_xor(mx, 16, 32));
    float s = 0.f;
    for (int cc = 0; cc < 128; cc++)
      s += __expf((float)psh[wv][row * Tdim + c0 + cc] - mx);
    s += __shfl_xor(s, 16, 32);
    float inv = 1.0f / s;
    for (int cc = 0; cc < 128; cc++) {
      float x = (float)psh[wv][row * Tdim + c0 + cc];
      psh[wv][row * Tdim + c0 + cc] = (_Float16)(__expf(x - mx) * inv);
    }
  }
#if HAVE_TDM
  __builtin_amdgcn_s_wait_tensorcnt(0);  // V tile resident (no-op for non-issuers)
#endif
  __syncthreads();
  v8f o0 = {}, o1 = {};  // AV: out[16 x 32] accumulated over K=256
#pragma unroll
  for (int kc = 0; kc < 8; kc++) {
    int kbase = kc * 32;
    v16h ap;
#pragma unroll
    for (int i = 0; i < 8; i++) {
      ap[i]     = psh[wv][l16 * Tdim + kbase + koff + i];
      ap[8 + i] = psh[wv][l16 * Tdim + kbase + 8 + koff + i];
    }
    v16h bv0, bv1;
#pragma unroll
    for (int e = 0; e < 16; e++) {
      bv0[e] = vsh[(kbase + kgrp + e) * Hdim + l16];
      bv1[e] = vsh[(kbase + kgrp + e) * Hdim + 16 + l16];
    }
    o0 = WMMA_F16(ap, bv0, o0);
    o1 = WMMA_F16(ap, bv1, o1);
  }
#pragma unroll
  for (int v = 0; v < 8; v++) {
    int tr = rows0 + v + voff;
    wvout[((size_t)b * Tdim + tr) * Hdim + l16] = o0[v];
    wvout[((size_t)b * Tdim + tr) * Hdim + 16 + l16] = o1[v];
  }
}

// ---------------------------------------------------------------------------
// Kernel 5: LN2 -> complex linear (16->64 x4) -> LN3 (over C, real/imag
// separately per the [N,2,F,T,C] transpose) -> PReLU -> +residual -> out.
// One wave per (b,t) row.
// ---------------------------------------------------------------------------
__global__ void k_final(const float* __restrict__ wvin,
                        const float* __restrict__ g2, const float* __restrict__ b2,
                        const float* __restrict__ Wr, const float* __restrict__ br,
                        const float* __restrict__ Wi, const float* __restrict__ bi,
                        const float* __restrict__ g3, const float* __restrict__ b3,
                        const float* __restrict__ prelu,
                        const float* __restrict__ inp, float* __restrict__ out) {
  int lane = threadIdx.x & 31, wv = threadIdx.x >> 5;
  size_t r = (size_t)blockIdx.x * 8 + wv;
  int b = (int)(r / Tdim), t = (int)(r % Tdim);
  int n = b / Fdim, f = b % Fdim;
  float y = wvin[r * Hdim + lane];
  float mean = wave_sum(y) * (1.f / Hdim);
  float d = y - mean;
  float yn = d * rsqrtf(wave_sum(d * d) * (1.f / Hdim) + 1e-5f) * g2[lane] + b2[lane];
  float xr[16], xi[16];
#pragma unroll
  for (int j = 0; j < 16; j++) {
    xr[j] = __shfl(yn, j, 32);
    xi[j] = __shfl(yn, j + 16, 32);
  }
  float orv[2], oiv[2];
#pragma unroll
  for (int u = 0; u < 2; u++) {
    int c = lane * 2 + u;
    float lrr = 0, lii = 0, lri = 0, lir = 0;
#pragma unroll
    for (int j = 0; j < 16; j++) {
      float wr = Wr[c * 16 + j], wi = Wi[c * 16 + j];
      lrr += xr[j] * wr;  lri += xi[j] * wr;
      lii += xi[j] * wi;  lir += xr[j] * wi;
    }
    orv[u] = (lrr + br[c]) - (lii + bi[c]);
    oiv[u] = (lri + br[c]) + (lir + bi[c]);
  }
  float mr = wave_sum(orv[0] + orv[1]) * (1.f / Cdim);
  float vr = wave_sum((orv[0] - mr) * (orv[0] - mr) + (orv[1] - mr) * (orv[1] - mr)) * (1.f / Cdim);
  float rsr = rsqrtf(vr + 1e-5f);
  float mi = wave_sum(oiv[0] + oiv[1]) * (1.f / Cdim);
  float vi = wave_sum((oiv[0] - mi) * (oiv[0] - mi) + (oiv[1] - mi) * (oiv[1] - mi)) * (1.f / Cdim);
  float rsi = rsqrtf(vi + 1e-5f);
  float a = prelu[0];
#pragma unroll
  for (int u = 0; u < 2; u++) {
    int c = lane * 2 + u;
    float vre = (orv[u] - mr) * rsr * g3[c] + b3[c];
    float vim = (oiv[u] - mi) * rsi * g3[c] + b3[c];
    vre = vre >= 0.f ? vre : a * vre;
    vim = vim >= 0.f ? vim : a * vim;
    size_t i0 = ((((size_t)n * Cdim + c) * Fdim + f) * Tdim + t) * 2;
    out[i0]     = vre + inp[i0];
    out[i0 + 1] = vim + inp[i0 + 1];
  }
}

// ---------------------------------------------------------------------------
extern "C" void kernel_launch(void* const* d_in, const int* in_sizes, int n_in,
                              void* d_out, int out_size, void* d_ws, size_t ws_size,
                              hipStream_t stream) {
  (void)in_sizes; (void)n_in; (void)out_size; (void)ws_size;
  const float* inputs = (const float*)d_in[0];
  const float* ln1_g = (const float*)d_in[1];
  const float* ln1_b = (const float*)d_in[2];
  const float* Wih_q = (const float*)d_in[3];
  const float* Whh_q = (const float*)d_in[4];
  const float* bih_q = (const float*)d_in[5];
  const float* bhh_q = (const float*)d_in[6];
  const float* Wih_k = (const float*)d_in[7];
  const float* Whh_k = (const float*)d_in[8];
  const float* bih_k = (const float*)d_in[9];
  const float* bhh_k = (const float*)d_in[10];
  const float* Wih_v = (const float*)d_in[11];
  const float* Whh_v = (const float*)d_in[12];
  const float* bih_v = (const float*)d_in[13];
  const float* bhh_v = (const float*)d_in[14];
  const float* ln2_g = (const float*)d_in[15];
  const float* ln2_b = (const float*)d_in[16];
  const float* Wr = (const float*)d_in[17];
  const float* br = (const float*)d_in[18];
  const float* Wi = (const float*)d_in[19];
  const float* bi = (const float*)d_in[20];
  const float* ln3_g = (const float*)d_in[21];
  const float* ln3_b = (const float*)d_in[22];
  const float* prelu_a = (const float*)d_in[23];
  float* out = (float*)d_out;

  char* ws = (char*)d_ws;
  auto alloc = [&](size_t bytes) -> char* {
    char* p = ws;
    ws += (bytes + 255) & ~(size_t)255;
    return p;
  };
  _Float16* wih16 = (_Float16*)alloc((size_t)3 * Ddim * Ddim * 2);
  _Float16* whh16 = (_Float16*)alloc((size_t)3 * Ddim * Hdim * 2);
  float*    bias  = (float*)   alloc((size_t)3 * Ddim * 4);
  _Float16* xn16  = (_Float16*)alloc(Mrows * Ddim * 2);
  float*    gpre  = (float*)   alloc((size_t)3 * Mrows * Ddim * 4);
  _Float16* qkv16 = (_Float16*)alloc((size_t)3 * Mrows * Hdim * 2);
  float*    wvb   = (float*)   alloc(Mrows * Hdim * 4);

  k_prep<<<dim3((3 * Ddim * Ddim + 255) / 256), 256, 0, stream>>>(
      Wih_q, Wih_k, Wih_v, Whh_q, Whh_k, Whh_v,
      bih_q, bhh_q, bih_k, bhh_k, bih_v, bhh_v, wih16, whh16, bias);
  k_ln1<<<dim3((unsigned)(Mrows / 8)), 256, 0, stream>>>(inputs, ln1_g, ln1_b, xn16);
  k_inproj<<<dim3((unsigned)(Mrows / 64), 3), 256, 0, stream>>>(xn16, wih16, bias, gpre);
  k_lstm<<<dim3(Bdim / 16, 3), 256, 0, stream>>>(gpre, whh16, qkv16);
  k_attn<<<dim3(Bdim * 4), 128, 0, stream>>>(qkv16, wvb);
  k_final<<<dim3((unsigned)(Mrows / 8)), 256, 0, stream>>>(
      wvb, ln2_g, ln2_b, Wr, br, Wi, bi, ln3_g, ln3_b, prelu_a, inputs, out);
}
